// DaViTChannelBlock_29137058136229
// MI455X (gfx1250) — matrix operand order repaired and checked by reading, not compiled, not run on hardware
//
#include <hip/hip_runtime.h>
#include <hip/hip_bf16.h>
#include <math.h>

// ---------------------------------------------------------------------------
// DaViT channel-attention block for MI455X (gfx1250), wave32 + WMMA bf16.
// GEMMs: v_wmma_f32_16x16x32_bf16.  Staging: global_load_async_to_lds_b128
// (ASYNCcnt) with a 3-stage LDS pipeline in the main GEMM.
// ---------------------------------------------------------------------------

typedef __bf16 bfv16 __attribute__((ext_vector_type(16)));
typedef float  f32v8 __attribute__((ext_vector_type(8)));

union Frag16 {
    bfv16 v;
    unsigned short u[16];
    unsigned int   w[8];
};

union Pack8 {
    uint4 v;
    unsigned short s[8];
};

__device__ __forceinline__ unsigned short f2bf(float f) {
    unsigned int u = __float_as_uint(f);
    u += 0x7FFFu + ((u >> 16) & 1u);     // round-to-nearest-even
    return (unsigned short)(u >> 16);
}

// ---- CDNA5 async global->LDS copy (tracked by ASYNCcnt) -------------------
__device__ __forceinline__ void async_b128(unsigned lds_addr, const void* gptr) {
    unsigned long long ga = (unsigned long long)(size_t)gptr;
    asm volatile("global_load_async_to_lds_b128 %0, %1, off"
                 :: "v"(lds_addr), "v"(ga) : "memory");
}
__device__ __forceinline__ void wait_async0() {
    asm volatile("s_wait_asynccnt 0x0" ::: "memory");
}
__device__ __forceinline__ void wait_async4() {
    asm volatile("s_wait_asynccnt 0x4" ::: "memory");
}

#define BB   32
#define HH   32
#define WW2  32
#define NN   1024          // H*W
#define CC   768
#define GG   8
#define HD   96            // CC/GG
#define CHID 3072          // 4*CC
#define BN   (BB * NN)     // 32768 rows

// ---------------------------------------------------------------------------
// Depthwise 3x3 conv (SAME) + bias + residual.  x:(B,N,C) channel-fastest.
// ---------------------------------------------------------------------------
__global__ __launch_bounds__(256) void dwconv_kernel(
    const float* __restrict__ x, const float* __restrict__ w,
    const float* __restrict__ bias, float* __restrict__ out)
{
    int c = blockIdx.x * 256 + threadIdx.x;      // gridDim.x = CC/256
    int n = blockIdx.y;                          // pixel
    int b = blockIdx.z;
    int y = n >> 5, xx = n & 31;
    const float* xb = x + (size_t)b * NN * CC;
    float acc = bias[c];
    #pragma unroll
    for (int dy = -1; dy <= 1; ++dy) {
        int yy = y + dy;
        if (yy < 0 || yy >= HH) continue;
        #pragma unroll
        for (int dx = -1; dx <= 1; ++dx) {
            int xc = xx + dx;
            if (xc < 0 || xc >= WW2) continue;
            acc += w[c * 9 + (dy + 1) * 3 + (dx + 1)] *
                   xb[(size_t)(yy * WW2 + xc) * CC + c];
        }
    }
    size_t idx = ((size_t)b * NN + n) * CC + c;
    out[idx] = x[idx] + acc;
}

// ---------------------------------------------------------------------------
// Row layernorm (C=768) -> bf16
// ---------------------------------------------------------------------------
__global__ __launch_bounds__(256) void layernorm_kernel(
    const float* __restrict__ x, const float* __restrict__ g,
    const float* __restrict__ b, unsigned short* __restrict__ out)
{
    __shared__ float s1[256], s2[256];
    int row = blockIdx.x, tid = threadIdx.x;
    const float* xr = x + (size_t)row * CC;
    float vals[3], lsum = 0.f, lsq = 0.f;
    #pragma unroll
    for (int i = 0; i < 3; ++i) {
        float v = xr[tid + i * 256];
        vals[i] = v; lsum += v; lsq += v * v;
    }
    s1[tid] = lsum; s2[tid] = lsq;
    __syncthreads();
    for (int off = 128; off > 0; off >>= 1) {
        if (tid < off) { s1[tid] += s1[tid + off]; s2[tid] += s2[tid + off]; }
        __syncthreads();
    }
    float mean = s1[0] * (1.0f / CC);
    float var  = s2[0] * (1.0f / CC) - mean * mean;
    float rstd = rsqrtf(var + 1e-5f);
    #pragma unroll
    for (int i = 0; i < 3; ++i) {
        int c = tid + i * 256;
        out[(size_t)row * CC + c] = f2bf((vals[i] - mean) * rstd * g[c] + b[c]);
    }
}

// ---------------------------------------------------------------------------
// fp32 -> bf16 conversion (weights)
// ---------------------------------------------------------------------------
__global__ void convert_f32_bf16_kernel(const float* __restrict__ s,
                                        unsigned short* __restrict__ d, int n)
{
    int i = blockIdx.x * 256 + threadIdx.x;
    if (i < n) d[i] = f2bf(s[i]);
}

// ---------------------------------------------------------------------------
// Tiled WMMA GEMM: out[m,n] = sum_k A[m,k] * W[n,k]  (+ epilogue)
//   A: (M x K) bf16 row-major, W: (Nn x K) bf16 row-major (torch weight layout)
//   block = 256 thr (8 waves), tile 128x128, K-step 32.
//   3-stage LDS pipeline fed by global_load_async_to_lds_b128 (ASYNCcnt).
//   EPI 0: bf16 out, +bias       (qkv)
//   EPI 1: bf16 out, gelu(x+b)   (fc1)
//   EPI 2: f32  out, x+b+res     (proj, fc2)
// ---------------------------------------------------------------------------
#define LDT 40   // LDS row stride (32 + 8 pad), multiple of 8

template <int EPI>
__global__ __launch_bounds__(256) void gemm_bf16_wmma(
    const unsigned short* __restrict__ A, int lda,
    const unsigned short* __restrict__ W, int ldw,
    const float* __restrict__ bias, const float* __restrict__ res,
    void* __restrict__ outv, int K)
{
    __shared__ __align__(16) unsigned short As[3][128 * LDT];
    __shared__ __align__(16) unsigned short Bs[3][128 * LDT];

    const int tid  = threadIdx.x;
    const int lane = tid & 31, wid = tid >> 5;
    const int half = lane >> 4, lm = lane & 15;
    const int wm = wid & 1;        // 2 wave-rows  (64 rows each)
    const int wn = wid >> 1;       // 4 wave-cols  (32 cols each)
    const int m0 = blockIdx.y * 128, n0 = blockIdx.x * 128;

    // per-thread staging chunk coordinates (2 chunks each for A and B)
    const int r0 = tid >> 2, c8 = (tid & 3) << 3;
    const int r1 = r0 + 64;

    f32v8 acc[4][2];
    const f32v8 z = {0.f, 0.f, 0.f, 0.f, 0.f, 0.f, 0.f, 0.f};
    #pragma unroll
    for (int i = 0; i < 4; ++i)
        #pragma unroll
        for (int j = 0; j < 2; ++j) acc[i][j] = z;

    const int nsteps = K >> 5;

    // prologue: async-stage k-tile 0 into buffer 0
    {
        async_b128((unsigned)(size_t)&As[0][r0 * LDT + c8],
                   A + (size_t)(m0 + r0) * lda + c8);
        async_b128((unsigned)(size_t)&As[0][r1 * LDT + c8],
                   A + (size_t)(m0 + r1) * lda + c8);
        async_b128((unsigned)(size_t)&Bs[0][r0 * LDT + c8],
                   W + (size_t)(n0 + r0) * ldw + c8);
        async_b128((unsigned)(size_t)&Bs[0][r1 * LDT + c8],
                   W + (size_t)(n0 + r1) * ldw + c8);
    }

    for (int s = 0; s < nsteps; ++s) {
        const int cur = s % 3;
        if (s + 1 < nsteps) {
            const int nxt = (s + 1) % 3;
            const int kk = (s + 1) << 5;
            async_b128((unsigned)(size_t)&As[nxt][r0 * LDT + c8],
                       A + (size_t)(m0 + r0) * lda + kk + c8);
            async_b128((unsigned)(size_t)&As[nxt][r1 * LDT + c8],
                       A + (size_t)(m0 + r1) * lda + kk + c8);
            async_b128((unsigned)(size_t)&Bs[nxt][r0 * LDT + c8],
                       W + (size_t)(n0 + r0) * ldw + kk + c8);
            async_b128((unsigned)(size_t)&Bs[nxt][r1 * LDT + c8],
                       W + (size_t)(n0 + r1) * ldw + kk + c8);
            wait_async4();             // in-order: tile `s` has landed
        } else {
            wait_async0();
        }
        __syncthreads();

        Frag16 af[4], bfr[2];
        #pragma unroll
        for (int i = 0; i < 4; ++i) {
            int m = wm * 64 + i * 16 + lm;
            #pragma unroll
            for (int p = 0; p < 8; ++p) {
                int k = ((p >> 2) << 4) + (half << 3) + ((p & 3) << 1);
                af[i].w[p] = *(const unsigned int*)&As[cur][m * LDT + k];
            }
        }
        #pragma unroll
        for (int j = 0; j < 2; ++j) {
            int n = wn * 32 + j * 16 + lm;
            #pragma unroll
            for (int p = 0; p < 8; ++p) {
                int k = ((p >> 2) << 4) + (half << 3) + ((p & 3) << 1);
                bfr[j].w[p] = *(const unsigned int*)&Bs[cur][n * LDT + k];
            }
        }
        #pragma unroll
        for (int i = 0; i < 4; ++i)
            #pragma unroll
            for (int j = 0; j < 2; ++j)
                acc[i][j] = __builtin_amdgcn_wmma_f32_16x16x32_bf16(
                    false, af[i].v, false, bfr[j].v, (short)0, acc[i][j],
                    false, false);
    }

    const int ldo = gridDim.x * 128;   // Nn
    #pragma unroll
    for (int i = 0; i < 4; ++i) {
        int mg = m0 + wm * 64 + i * 16 + half * 8;
        #pragma unroll
        for (int j = 0; j < 2; ++j) {
            int ng = n0 + wn * 32 + j * 16 + lm;
            float bv = bias[ng];
            #pragma unroll
            for (int v = 0; v < 8; ++v) {
                size_t oidx = (size_t)(mg + v) * ldo + ng;
                float val = acc[i][j][v] + bv;
                if constexpr (EPI == 0) {
                    ((unsigned short*)outv)[oidx] = f2bf(val);
                } else if constexpr (EPI == 1) {
                    float ge = 0.5f * val * (1.0f + erff(val * 0.70710678f));
                    ((unsigned short*)outv)[oidx] = f2bf(ge);
                } else {
                    ((float*)outv)[oidx] = val + res[oidx];
                }
            }
        }
    }
}

// ---------------------------------------------------------------------------
// Channel-attention logits: per (b,g), S = scale * Q^T K over N=1024, then
// row softmax.  Q/K staged TRANSPOSED in LDS (qsT[c][n], ksT[d][n]) so that
// WMMA fragment k-pairs are contiguous dwords.
// block = 128 thr (4 waves, each 48x48 = 3x3 WMMA tiles)
// ---------------------------------------------------------------------------
#define LDS_T 34   // transposed stride: 32 n-values + 2 pad (even)

__global__ __launch_bounds__(128) void attn_logits_kernel(
    const unsigned short* __restrict__ qkv,
    unsigned short* __restrict__ attn, float scale)
{
    __shared__ __align__(16) unsigned short qsT[96 * LDS_T];
    __shared__ __align__(16) unsigned short ksT[96 * LDS_T];
    __shared__ float sS[96 * 97];

    const int z = blockIdx.x;
    const int b = z >> 3, g = z & 7;
    const int tid = threadIdx.x, lane = tid & 31, wid = tid >> 5;
    const int half = lane >> 4, lm = lane & 15;
    const int wm = wid & 1, wn = wid >> 1;

    const unsigned short* base = qkv + (size_t)b * NN * (3 * CC) + g * HD;

    f32v8 acc[3][3];
    const f32v8 zz = {0.f, 0.f, 0.f, 0.f, 0.f, 0.f, 0.f, 0.f};
    #pragma unroll
    for (int i = 0; i < 3; ++i)
        #pragma unroll
        for (int j = 0; j < 3; ++j) acc[i][j] = zz;

    for (int nn = 0; nn < NN; nn += 32) {
        // stage 32 rows x 96 cols of Q and K, transposed, row-pairs packed
        for (int ch = tid; ch < 192; ch += 128) {   // 12 c-chunks x 16 row-pairs
            int c8 = (ch >> 4) << 3;
            int r  = (ch & 15) << 1;
            const unsigned short* q0 = base + (size_t)(nn + r) * (3 * CC);
            const unsigned short* q1 = q0 + 3 * CC;
            Pack8 a0, a1, b0, b1;
            a0.v = *(const uint4*)(q0 + c8);
            a1.v = *(const uint4*)(q1 + c8);
            b0.v = *(const uint4*)(q0 + CC + c8);
            b1.v = *(const uint4*)(q1 + CC + c8);
            #pragma unroll
            for (int e = 0; e < 8; ++e) {
                unsigned int qa = (unsigned int)a0.s[e] |
                                  ((unsigned int)a1.s[e] << 16);
                unsigned int kb = (unsigned int)b0.s[e] |
                                  ((unsigned int)b1.s[e] << 16);
                *(unsigned int*)&qsT[(c8 + e) * LDS_T + r] = qa;
                *(unsigned int*)&ksT[(c8 + e) * LDS_T + r] = kb;
            }
        }
        __syncthreads();

        Frag16 af[3], bfr[3];
        #pragma unroll
        for (int i = 0; i < 3; ++i) {          // A = Q^T : m=c, k=n-row
            int m = wm * 48 + i * 16 + lm;
            #pragma unroll
            for (int p = 0; p < 8; ++p) {
                int k = ((p >> 2) << 4) + (half << 3) + ((p & 3) << 1);
                af[i].w[p] = *(const unsigned int*)&qsT[m * LDS_T + k];
            }
        }
        #pragma unroll
        for (int jj = 0; jj < 3; ++jj) {       // B = K : k=n-row, n=d
            int n = wn * 48 + jj * 16 + lm;
            #pragma unroll
            for (int p = 0; p < 8; ++p) {
                int k = ((p >> 2) << 4) + (half << 3) + ((p & 3) << 1);
                bfr[jj].w[p] = *(const unsigned int*)&ksT[n * LDS_T + k];
            }
        }
        #pragma unroll
        for (int i = 0; i < 3; ++i)
            #pragma unroll
            for (int j = 0; j < 3; ++j)
                acc[i][j] = __builtin_amdgcn_wmma_f32_16x16x32_bf16(
                    false, af[i].v, false, bfr[j].v, (short)0, acc[i][j],
                    false, false);
        __syncthreads();
    }

    #pragma unroll
    for (int i = 0; i < 3; ++i)
        #pragma unroll
        for (int j = 0; j < 3; ++j)
            #pragma unroll
            for (int v = 0; v < 8; ++v)
                sS[(wm * 48 + i * 16 + half * 8 + v) * 97 +
                   wn * 48 + j * 16 + lm] = acc[i][j][v] * scale;
    __syncthreads();

    if (tid < HD) {
        float mx = -3.0e38f;
        for (int d = 0; d < HD; ++d) mx = fmaxf(mx, sS[tid * 97 + d]);
        float s = 0.f;
        for (int d = 0; d < HD; ++d) {
            float e = __expf(sS[tid * 97 + d] - mx);
            sS[tid * 97 + d] = e; s += e;
        }
        float inv = 1.0f / s;
        unsigned short* arow = attn + (size_t)z * HD * HD + tid * HD;
        for (int d = 0; d < HD; ++d) arow[d] = f2bf(sS[tid * 97 + d] * inv);
    }
}

// ---------------------------------------------------------------------------
// Apply attention: per (b,g): ao[n,c] = sum_d attn[c,d] * V[n,d]
// block = 256 thr (8 waves: 4x2), tile 128(n) x 96(c), K = 96 (3 steps)
// Staging via async global->LDS copies.
// ---------------------------------------------------------------------------
__global__ __launch_bounds__(256) void apply_attn_kernel(
    const unsigned short* __restrict__ qkv,
    const unsigned short* __restrict__ attn,
    unsigned short* __restrict__ ao)
{
    __shared__ __align__(16) unsigned short Vs[128 * 104];
    __shared__ __align__(16) unsigned short Ts[96 * 104];

    const int z = blockIdx.y;
    const int b = z >> 3, g = z & 7;
    const int n0 = blockIdx.x * 128;
    const int tid = threadIdx.x, lane = tid & 31, wid = tid >> 5;
    const int half = lane >> 4, lm = lane & 15;
    const int wm = wid & 3, wn = wid >> 2;   // 4 row-waves x 2 col-waves

    #pragma unroll
    for (int t = 0; t < 6; ++t) {            // V tile: 1536 chunks
        int ch = tid + t * 256;
        int r = ch / 12, c8 = (ch % 12) * 8;
        async_b128((unsigned)(size_t)&Vs[r * 104 + c8],
            qkv + (size_t)(b * NN + n0 + r) * (3 * CC) + 2 * CC + g * HD + c8);
    }
    for (int ch = tid; ch < 1152; ch += 256) {  // attn: 96x96
        int r = ch / 12, c8 = (ch % 12) * 8;
        async_b128((unsigned)(size_t)&Ts[r * 104 + c8],
                   attn + (size_t)z * HD * HD + r * HD + c8);
    }
    wait_async0();
    __syncthreads();

    f32v8 acc[2][3];
    const f32v8 zz = {0.f, 0.f, 0.f, 0.f, 0.f, 0.f, 0.f, 0.f};
    #pragma unroll
    for (int i = 0; i < 2; ++i)
        #pragma unroll
        for (int j = 0; j < 3; ++j) acc[i][j] = zz;

    #pragma unroll
    for (int kk = 0; kk < HD; kk += 32) {
        Frag16 af[2], bfr[3];
        #pragma unroll
        for (int i = 0; i < 2; ++i) {
            int m = wm * 32 + i * 16 + lm;
            #pragma unroll
            for (int p = 0; p < 8; ++p) {
                int k = kk + ((p >> 2) << 4) + (half << 3) + ((p & 3) << 1);
                af[i].w[p] = *(const unsigned int*)&Vs[m * 104 + k];
            }
        }
        #pragma unroll
        for (int j = 0; j < 3; ++j) {
            int n = wn * 48 + j * 16 + lm;
            #pragma unroll
            for (int p = 0; p < 8; ++p) {
                int k = kk + ((p >> 2) << 4) + (half << 3) + ((p & 3) << 1);
                bfr[j].w[p] = *(const unsigned int*)&Ts[n * 104 + k];
            }
        }
        #pragma unroll
        for (int i = 0; i < 2; ++i)
            #pragma unroll
            for (int j = 0; j < 3; ++j)
                acc[i][j] = __builtin_amdgcn_wmma_f32_16x16x32_bf16(
                    false, af[i].v, false, bfr[j].v, (short)0, acc[i][j],
                    false, false);
    }

    #pragma unroll
    for (int i = 0; i < 2; ++i) {
        int mg = n0 + wm * 32 + i * 16 + half * 8;
        #pragma unroll
        for (int j = 0; j < 3; ++j) {
            int cg = wn * 48 + j * 16 + lm;
            #pragma unroll
            for (int v = 0; v < 8; ++v)
                ao[(size_t)(b * NN + mg + v) * CC + g * HD + cg] =
                    f2bf(acc[i][j][v]);
        }
    }
}

// ---------------------------------------------------------------------------
// Launch.  Workspace layout (bytes), ~477 MB total:
//   x1   f32  (BN*C)      @ 0
//   x2   f32  (BN*C)      @ 100663296
//   xn   bf16 (BN*C)      @ 201326592   (reused: LN out / attn out)
//   qkvh bf16 (BN*4C)     @ 251658240   (reused: qkv then mlp hidden)
//   attn bf16 (B*G*96*96) @ 452984832
//   wb   bf16 weights     @ 462422016
// ---------------------------------------------------------------------------
extern "C" void kernel_launch(void* const* d_in, const int* in_sizes, int n_in,
                              void* d_out, int out_size, void* d_ws,
                              size_t ws_size, hipStream_t stream)
{
    const float* x      = (const float*)d_in[0];
    const float* dw1_w  = (const float*)d_in[3];
    const float* dw1_b  = (const float*)d_in[4];
    const float* ln1_g  = (const float*)d_in[5];
    const float* ln1_b  = (const float*)d_in[6];
    const float* qkv_w  = (const float*)d_in[7];
    const float* qkv_b  = (const float*)d_in[8];
    const float* proj_w = (const float*)d_in[9];
    const float* proj_b = (const float*)d_in[10];
    const float* dw2_w  = (const float*)d_in[11];
    const float* dw2_b  = (const float*)d_in[12];
    const float* ln2_g  = (const float*)d_in[13];
    const float* ln2_b  = (const float*)d_in[14];
    const float* fc1_w  = (const float*)d_in[15];
    const float* fc1_b  = (const float*)d_in[16];
    const float* fc2_w  = (const float*)d_in[17];
    const float* fc2_b  = (const float*)d_in[18];

    char* ws = (char*)d_ws;
    float*          x1   = (float*)(ws);
    float*          x2   = (float*)(ws + 100663296ull);
    unsigned short* xn   = (unsigned short*)(ws + 201326592ull);
    unsigned short* qkvh = (unsigned short*)(ws + 251658240ull);
    unsigned short* attn = (unsigned short*)(ws + 452984832ull);
    unsigned short* wqkv = (unsigned short*)(ws + 462422016ull);
    unsigned short* wprj = wqkv + (size_t)3 * CC * CC;
    unsigned short* wfc1 = wprj + (size_t)CC * CC;
    unsigned short* wfc2 = wfc1 + (size_t)CHID * CC;

    auto cvt = [&](const float* s, unsigned short* d, int n) {
        convert_f32_bf16_kernel<<<(n + 255) / 256, 256, 0, stream>>>(s, d, n);
    };
    cvt(qkv_w, wqkv, 3 * CC * CC);
    cvt(proj_w, wprj, CC * CC);
    cvt(fc1_w, wfc1, CHID * CC);
    cvt(fc2_w, wfc2, CC * CHID);

    dim3 dwgrid(CC / 256, NN, BB);
    // x1 = x + dwconv1(x)
    dwconv_kernel<<<dwgrid, 256, 0, stream>>>(x, dw1_w, dw1_b, x1);
    // xn = LN1(x1) (bf16)
    layernorm_kernel<<<BN, 256, 0, stream>>>(x1, ln1_g, ln1_b, xn);
    // qkv = xn @ qkv_w^T + qkv_b    (bf16 out)
    gemm_bf16_wmma<0><<<dim3(3 * CC / 128, BN / 128), 256, 0, stream>>>(
        xn, CC, wqkv, CC, qkv_b, nullptr, qkvh, CC);
    // attn = softmax(scale * Q^T K) per (b,g)
    attn_logits_kernel<<<BB * GG, 128, 0, stream>>>(qkvh, attn, 0.03125f);
    // xn = attn . V   (reuse xn buffer as attention output, bf16)
    apply_attn_kernel<<<dim3(NN / 128, BB * GG), 256, 0, stream>>>(qkvh, attn, xn);
    // x2 = x1 + xn @ proj_w^T + proj_b   (f32)
    gemm_bf16_wmma<2><<<dim3(CC / 128, BN / 128), 256, 0, stream>>>(
        xn, CC, wprj, CC, proj_b, x1, x2, CC);
    // x3 (in x1 buffer) = x2 + dwconv2(x2)
    dwconv_kernel<<<dwgrid, 256, 0, stream>>>(x2, dw2_w, dw2_b, x1);
    // xn = LN2(x3)
    layernorm_kernel<<<BN, 256, 0, stream>>>(x1, ln2_g, ln2_b, xn);
    // h (in qkvh buffer) = gelu(xn @ fc1_w^T + fc1_b)   (bf16)
    gemm_bf16_wmma<1><<<dim3(CHID / 128, BN / 128), 256, 0, stream>>>(
        xn, CC, wfc1, CC, fc1_b, nullptr, qkvh, CC);
    // out = x3 + h @ fc2_w^T + fc2_b   (f32 -> d_out)
    gemm_bf16_wmma<2><<<dim3(CC / 128, BN / 128), 256, 0, stream>>>(
        qkvh, CHID, wfc2, CHID, fc2_b, x1, (float*)d_out, CHID);
}
